// MambaForSequenceClassification_74208444940509
// MI455X (gfx1250) — matrix operand order repaired and checked by reading, not compiled, not run on hardware
//
#include <hip/hip_runtime.h>
#include <hip/hip_bf16.h>
#include <math.h>

#define BATCH 8
#define SEQ 1024
#define DMODEL 768
#define DINNER 1536
#define DSTATE 16
#define DTRANK 48
#define DCONV 4
#define MROWS (BATCH * SEQ)   // 8192

typedef __attribute__((ext_vector_type(16))) _Float16 v16h;
typedef __attribute__((ext_vector_type(8)))  _Float16 v8h;
typedef __attribute__((ext_vector_type(8)))  float    v8f;

__device__ __forceinline__ float siluf(float x)     { return x / (1.0f + __expf(-x)); }
__device__ __forceinline__ float softplusf(float x) { return x > 20.0f ? x : log1pf(__expf(x)); }

// ---------------------------------------------------------------------------
// f32 -> f16 conversion (weight staging)
// ---------------------------------------------------------------------------
__global__ void cvt_f16_kernel(const float* __restrict__ x, _Float16* __restrict__ o, int n) {
    int i = blockIdx.x * blockDim.x + threadIdx.x;
    if (i < n) o[i] = (_Float16)x[i];
}

// ---------------------------------------------------------------------------
// Embedding gather: h[b,s,:] = embed[ids[b,s],:]
// ---------------------------------------------------------------------------
__global__ void embed_kernel(const int* __restrict__ ids,
                             const float* __restrict__ embed,
                             float* __restrict__ h, int total) {
    int i = blockIdx.x * blockDim.x + threadIdx.x;
    if (i >= total) return;
    int dm = i % DMODEL;
    int bs = i / DMODEL;
    h[i] = embed[(size_t)ids[bs] * DMODEL + dm];
}

// ---------------------------------------------------------------------------
// RMSNorm over last dim (DMODEL), one block per row; writes f16 for GEMM-A
// ---------------------------------------------------------------------------
__global__ __launch_bounds__(256) void rmsnorm_f16_kernel(const float* __restrict__ x,
                                                          const float* __restrict__ w,
                                                          _Float16* __restrict__ o) {
    __shared__ float red[256];
    int row = blockIdx.x;
    const float* xr = x + (size_t)row * DMODEL;
    float s = 0.f;
    for (int i = threadIdx.x; i < DMODEL; i += 256) { float v = xr[i]; s += v * v; }
    red[threadIdx.x] = s;
    __syncthreads();
    for (int off = 128; off > 0; off >>= 1) {
        if (threadIdx.x < off) red[threadIdx.x] += red[threadIdx.x + off];
        __syncthreads();
    }
    float scale = rsqrtf(red[0] / (float)DMODEL + 1e-5f);
    for (int i = threadIdx.x; i < DMODEL; i += 256)
        o[(size_t)row * DMODEL + i] = (_Float16)(xr[i] * scale * w[i]);
}

// f32-output variant for the final norm (pooling reads f32)
__global__ __launch_bounds__(256) void rmsnorm_f32_kernel(const float* __restrict__ x,
                                                          const float* __restrict__ w,
                                                          float* __restrict__ o) {
    __shared__ float red[256];
    int row = blockIdx.x;
    const float* xr = x + (size_t)row * DMODEL;
    float s = 0.f;
    for (int i = threadIdx.x; i < DMODEL; i += 256) { float v = xr[i]; s += v * v; }
    red[threadIdx.x] = s;
    __syncthreads();
    for (int off = 128; off > 0; off >>= 1) {
        if (threadIdx.x < off) red[threadIdx.x] += red[threadIdx.x + off];
        __syncthreads();
    }
    float scale = rsqrtf(red[0] / (float)DMODEL + 1e-5f);
    for (int i = threadIdx.x; i < DMODEL; i += 256)
        o[(size_t)row * DMODEL + i] = xr[i] * scale * w[i];
}

// ---------------------------------------------------------------------------
// WMMA GEMM: C[M,N] = A[M,K] * W[N,K]^T  (+ epilogue), f16 inputs / f32 accum.
// One wave -> 32x32 output (2x2 WMMA tiles): A/B fragments each reused twice,
// 8x 16B loads per 4 WMMAs, no conversions in the loop.
// EPI: 0 none, 1 softplus(val + bias[n]), 2 val + resid[m,n]
// OUTH: 1 -> store f16, 0 -> store f32
// K tail supports rem == 16 (dt_proj K=48); all shapes here are mult-of-16.
// ---------------------------------------------------------------------------
template <int EPI, int OUTH>
__global__ __launch_bounds__(128) void gemm_wmma_kernel(
    const _Float16* __restrict__ A, int lda,
    const _Float16* __restrict__ W, int ldb,
    void* __restrict__ Cv, int ldc,
    int M, int N, int K,
    const float* __restrict__ bias,
    const float* __restrict__ resid, int ldr) {

    int wave = threadIdx.x >> 5;
    int lane = threadIdx.x & 31;
    int ntn  = (N + 31) >> 5;                 // n-groups of 32 columns
    int ngroups = (M >> 5) * ntn;
    int g = blockIdx.x * 4 + wave;
    if (g >= ngroups) return;                 // wave-uniform exit

    int n0 = (g % ntn) << 5;
    int m0 = (g / ntn) << 5;
    bool n1v = (n0 + 16) < N;                 // second n-tile valid (N mult of 16)
    int n1 = n1v ? (n0 + 16) : n0;
    int r  = lane & 15;
    int kh = lane >> 4;

    const _Float16* a0p = A + (size_t)(m0 + r)      * lda;
    const _Float16* a1p = A + (size_t)(m0 + 16 + r) * lda;
    const _Float16* b0p = W + (size_t)(n0 + r)      * ldb;
    const _Float16* b1p = W + (size_t)(n1 + r)      * ldb;

    union Frag { v16h v; v8h h8[2]; };
    union Acc  { v8f v; float f[8]; } acc00 = {}, acc01 = {}, acc10 = {}, acc11 = {};

    int kfull = K & ~31;
    for (int k0 = 0; k0 < kfull; k0 += 32) {
        Frag fa0, fa1, fb0, fb1;
        // A frag: e0..7 = K[k0+kh*8 .. +7], e8..15 = K[k0+16+kh*8 .. +7]
        fa0.h8[0] = *(const v8h*)(a0p + k0 + kh * 8);
        fa0.h8[1] = *(const v8h*)(a0p + k0 + 16 + kh * 8);
        fa1.h8[0] = *(const v8h*)(a1p + k0 + kh * 8);
        fa1.h8[1] = *(const v8h*)(a1p + k0 + 16 + kh * 8);
        // B frag: same K layout as A: e0..7 = K[k0+kh*8 .. +7], e8..15 = K[k0+16+kh*8 .. +7]
        fb0.h8[0] = *(const v8h*)(b0p + k0 + kh * 8);
        fb0.h8[1] = *(const v8h*)(b0p + k0 + 16 + kh * 8);
        fb1.h8[0] = *(const v8h*)(b1p + k0 + kh * 8);
        fb1.h8[1] = *(const v8h*)(b1p + k0 + 16 + kh * 8);

        acc00.v = __builtin_amdgcn_wmma_f32_16x16x32_f16(false, fa0.v, false, fb0.v, (short)0, acc00.v, false, false);
        acc01.v = __builtin_amdgcn_wmma_f32_16x16x32_f16(false, fa0.v, false, fb1.v, (short)0, acc01.v, false, false);
        acc10.v = __builtin_amdgcn_wmma_f32_16x16x32_f16(false, fa1.v, false, fb0.v, (short)0, acc10.v, false, false);
        acc11.v = __builtin_amdgcn_wmma_f32_16x16x32_f16(false, fa1.v, false, fb1.v, (short)0, acc11.v, false, false);
    }

    if (kfull < K) {  // rem == 16: A and B high halves (e8..15) are all >= K -> zero
        Frag fa0 = {}, fa1 = {}, fb0 = {}, fb1 = {};
        fa0.h8[0] = *(const v8h*)(a0p + kfull + kh * 8);
        fa1.h8[0] = *(const v8h*)(a1p + kfull + kh * 8);
        fb0.h8[0] = *(const v8h*)(b0p + kfull + kh * 8);
        fb1.h8[0] = *(const v8h*)(b1p + kfull + kh * 8);
        acc00.v = __builtin_amdgcn_wmma_f32_16x16x32_f16(false, fa0.v, false, fb0.v, (short)0, acc00.v, false, false);
        acc01.v = __builtin_amdgcn_wmma_f32_16x16x32_f16(false, fa0.v, false, fb1.v, (short)0, acc01.v, false, false);
        acc10.v = __builtin_amdgcn_wmma_f32_16x16x32_f16(false, fa1.v, false, fb0.v, (short)0, acc10.v, false, false);
        acc11.v = __builtin_amdgcn_wmma_f32_16x16x32_f16(false, fa1.v, false, fb1.v, (short)0, acc11.v, false, false);
    }

    // C/D layout: lanes 0-15 -> M=v, N=lane; lanes 16-31 -> M=8+v, N=lane-16
    int col = lane & 15;
    int mb  = (lane >> 4) << 3;
    auto store_tile = [&](Acc& a, int mt, int nt, bool valid) {
        if (!valid) return;
        #pragma unroll
        for (int v2 = 0; v2 < 8; ++v2) {
            int m = mt + mb + v2;
            int n = nt + col;
            float val = a.f[v2];
            if (EPI == 1) val = softplusf(val + bias[n]);
            if (EPI == 2) val += resid[(size_t)m * ldr + n];
            if (OUTH) ((_Float16*)Cv)[(size_t)m * ldc + n] = (_Float16)val;
            else      ((float*)Cv)[(size_t)m * ldc + n] = val;
        }
    };
    store_tile(acc00, m0,      n0, true);
    store_tile(acc01, m0,      n1, n1v);
    store_tile(acc10, m0 + 16, n0, true);
    store_tile(acc11, m0 + 16, n1, n1v);
}

// ---------------------------------------------------------------------------
// Depthwise causal conv (width 4) + bias + SiLU; input = xz[..., 0:DINNER] f32,
// output xi in f16 (GEMM-A for x_proj, and scan input).
// ---------------------------------------------------------------------------
__global__ void conv_silu_kernel(const float* __restrict__ xz,
                                 const float* __restrict__ cw,
                                 const float* __restrict__ cb,
                                 _Float16* __restrict__ xi, int total) {
    int i = blockIdx.x * blockDim.x + threadIdx.x;
    if (i >= total) return;
    int d = i % DINNER;
    int s = (i / DINNER) % SEQ;
    int b = i / (DINNER * SEQ);
    const float* base = xz + (size_t)b * SEQ * (2 * DINNER) + d;
    float acc = cb[d];
    #pragma unroll
    for (int t = 0; t < DCONV; ++t) {
        int si = s + t - (DCONV - 1);
        if (si >= 0) acc += cw[d * DCONV + t] * base[(size_t)si * (2 * DINNER)];
    }
    xi[i] = (_Float16)siluf(acc);
}

// ---------------------------------------------------------------------------
// Selective scan: thread = one (b,d) channel; h[16] in registers.
// B/C (32 halves per timestep) staged through LDS (as f32) in chunks of 8.
// Fused: +xi*D skip, * silu(z) gate; writes f16 (GEMM-A for out_proj).
// ---------------------------------------------------------------------------
__global__ __launch_bounds__(256) void scan_kernel(
    const _Float16* __restrict__ xdbc,  // [B,S,80] f16, B at +48, C at +64
    const float* __restrict__ dt,       // [B,S,DINNER] f32
    const _Float16* __restrict__ xi,    // [B,S,DINNER] f16
    const float* __restrict__ xz,       // [B,S,2*DINNER] f32, z at +DINNER
    const float* __restrict__ A_log,    // [DINNER,16]
    const float* __restrict__ Dp,       // [DINNER]
    _Float16* __restrict__ y) {         // [B,S,DINNER] f16
    __shared__ float sBC[8][32];
    int d = blockIdx.x * 256 + threadIdx.x;
    int b = blockIdx.y;

    float an[DSTATE];
    #pragma unroll
    for (int n = 0; n < DSTATE; ++n) an[n] = -__expf(A_log[(size_t)d * DSTATE + n]);
    float Dv = Dp[d];
    float hst[DSTATE];
    #pragma unroll
    for (int n = 0; n < DSTATE; ++n) hst[n] = 0.0f;

    for (int sc = 0; sc < SEQ / 8; ++sc) {
        __syncthreads();
        {   // 256 threads load 8 timesteps x 32 values of B|C
            int sl = threadIdx.x >> 5;
            int e  = threadIdx.x & 31;
            sBC[sl][e] = (float)xdbc[((size_t)b * SEQ + sc * 8 + sl) * 80 + DTRANK + e];
        }
        __syncthreads();
        #pragma unroll 1
        for (int ss = 0; ss < 8; ++ss) {
            int s = sc * 8 + ss;
            size_t base = ((size_t)b * SEQ + s) * DINNER + d;
            float dtv = dt[base];
            float xv  = (float)xi[base];
            float dx  = dtv * xv;
            float yv  = 0.0f;
            #pragma unroll
            for (int n = 0; n < DSTATE; ++n) {
                float dA = __expf(dtv * an[n]);
                hst[n] = dA * hst[n] + dx * sBC[ss][n];
                yv += hst[n] * sBC[ss][16 + n];
            }
            yv += xv * Dv;
            float zv = xz[((size_t)b * SEQ + s) * (2 * DINNER) + DINNER + d];
            y[base] = (_Float16)(yv * siluf(zv));
        }
    }
}

// ---------------------------------------------------------------------------
// Masked mean pool over sequence
// ---------------------------------------------------------------------------
__global__ void pool_kernel(const float* __restrict__ hn,
                            const int* __restrict__ mask,
                            float* __restrict__ pooled) {
    int i = blockIdx.x * blockDim.x + threadIdx.x;
    if (i >= BATCH * DMODEL) return;
    int b = i / DMODEL, dm = i % DMODEL;
    float s = 0.f, ms = 0.f;
    for (int t = 0; t < SEQ; ++t) {
        float m = (float)mask[b * SEQ + t];
        s  += m * hn[((size_t)b * SEQ + t) * DMODEL + dm];
        ms += m;
    }
    pooled[i] = s / fmaxf(ms, 1.0f);
}

// ---------------------------------------------------------------------------
// Classifier head: logits[b,l] = pooled[b] . cls_w[l] + cls_b[l]
// ---------------------------------------------------------------------------
__global__ void cls_kernel(const float* __restrict__ pooled,
                           const float* __restrict__ cw,
                           const float* __restrict__ cb,
                           float* __restrict__ out) {
    int o = threadIdx.x;
    if (o >= BATCH * 2) return;
    int b = o >> 1, l = o & 1;
    float s = cb[l];
    for (int i = 0; i < DMODEL; ++i) s += pooled[b * DMODEL + i] * cw[l * DMODEL + i];
    out[o] = s;
}

// ---------------------------------------------------------------------------
static inline int gemm_blocks(int M, int N) {
    int ngroups = (M >> 5) * ((N + 31) >> 5);
    return (ngroups + 3) / 4;
}

extern "C" void kernel_launch(void* const* d_in, const int* in_sizes, int n_in,
                              void* d_out, int out_size, void* d_ws, size_t ws_size,
                              hipStream_t stream) {
    const int*   ids       = (const int*)d_in[0];
    const int*   mask      = (const int*)d_in[1];
    const float* embed     = (const float*)d_in[2];
    const float* norm_w    = (const float*)d_in[3];
    const float* in_proj_w = (const float*)d_in[4];
    const float* conv_w    = (const float*)d_in[5];
    const float* conv_b    = (const float*)d_in[6];
    const float* x_proj_w  = (const float*)d_in[7];
    const float* dt_proj_w = (const float*)d_in[8];
    const float* dt_proj_b = (const float*)d_in[9];
    const float* A_log     = (const float*)d_in[10];
    const float* D_param   = (const float*)d_in[11];
    const float* out_pw    = (const float*)d_in[12];
    const float* norm_f_w  = (const float*)d_in[13];
    const float* cls_w     = (const float*)d_in[14];
    const float* cls_b     = (const float*)d_in[15];
    float* logits = (float*)d_out;

    // Workspace bump allocator (256B aligned)
    char* wp = (char*)d_ws;
    auto alloc = [&](size_t bytes) -> char* {
        char* r = wp; wp += (bytes + 255) & ~(size_t)255; return r;
    };
    float*     h      = (float*)    alloc((size_t)MROWS * DMODEL * 4);        // residual stream
    float*     hn     = (float*)    alloc((size_t)MROWS * DMODEL * 4);        // final norm out
    _Float16*  xnh    = (_Float16*) alloc((size_t)MROWS * DMODEL * 2);        // rmsnorm f16
    float*     xz     = (float*)    alloc((size_t)MROWS * 2 * DINNER * 4);    // in_proj out
    _Float16*  xih    = (_Float16*) alloc((size_t)MROWS * DINNER * 2);        // conv+silu f16
    _Float16*  xdbch  = (_Float16*) alloc((size_t)MROWS * 80 * 2);            // x_proj out f16
    float*     dtb    = (float*)    alloc((size_t)MROWS * DINNER * 4);        // softplus dt
    _Float16*  ybufh  = (_Float16*) alloc((size_t)MROWS * DINNER * 2);        // scan out f16
    float*     pooled = (float*)    alloc((size_t)BATCH * DMODEL * 4);
    _Float16*  w_ip   = (_Float16*) alloc((size_t)2 * DINNER * DMODEL * 2);   // f16 weights
    _Float16*  w_xp   = (_Float16*) alloc((size_t)(DTRANK + 2 * DSTATE) * DINNER * 2);
    _Float16*  w_dt   = (_Float16*) alloc((size_t)DINNER * DTRANK * 2);
    _Float16*  w_op   = (_Float16*) alloc((size_t)DMODEL * DINNER * 2);

    // 1) Embedding gather
    {
        int total = MROWS * DMODEL;
        embed_kernel<<<(total + 255) / 256, 256, 0, stream>>>(ids, embed, h, total);
    }

    // 2) Mamba layers
    for (int l = 0; l < 4; ++l) {
        const float* nw  = norm_w    + (size_t)l * DMODEL;
        const float* ipw = in_proj_w + (size_t)l * 2 * DINNER * DMODEL;
        const float* cw  = conv_w    + (size_t)l * DINNER * DCONV;
        const float* cb  = conv_b    + (size_t)l * DINNER;
        const float* xpw = x_proj_w  + (size_t)l * (DTRANK + 2 * DSTATE) * DINNER;
        const float* dpw = dt_proj_w + (size_t)l * DINNER * DTRANK;
        const float* dpb = dt_proj_b + (size_t)l * DINNER;
        const float* al  = A_log     + (size_t)l * DINNER * DSTATE;
        const float* dp  = D_param   + (size_t)l * DINNER;
        const float* opw = out_pw    + (size_t)l * DMODEL * DINNER;

        // stage this layer's weights as f16
        {
            int n1 = 2 * DINNER * DMODEL;
            int n2 = (DTRANK + 2 * DSTATE) * DINNER;
            int n3 = DINNER * DTRANK;
            int n4 = DMODEL * DINNER;
            cvt_f16_kernel<<<(n1 + 255) / 256, 256, 0, stream>>>(ipw, w_ip, n1);
            cvt_f16_kernel<<<(n2 + 255) / 256, 256, 0, stream>>>(xpw, w_xp, n2);
            cvt_f16_kernel<<<(n3 + 255) / 256, 256, 0, stream>>>(dpw, w_dt, n3);
            cvt_f16_kernel<<<(n4 + 255) / 256, 256, 0, stream>>>(opw, w_op, n4);
        }

        rmsnorm_f16_kernel<<<MROWS, 256, 0, stream>>>(h, nw, xnh);

        // in_proj: xz[M,3072](f32) = xnh[M,768] * w_ip[3072,768]^T
        gemm_wmma_kernel<0, 0><<<gemm_blocks(MROWS, 2 * DINNER), 128, 0, stream>>>(
            xnh, DMODEL, w_ip, DMODEL, xz, 2 * DINNER,
            MROWS, 2 * DINNER, DMODEL, nullptr, nullptr, 0);

        // conv + SiLU -> xih (f16)
        {
            int total = MROWS * DINNER;
            conv_silu_kernel<<<(total + 255) / 256, 256, 0, stream>>>(xz, cw, cb, xih, total);
        }

        // x_proj: xdbch[M,80](f16) = xih[M,1536] * w_xp[80,1536]^T
        gemm_wmma_kernel<0, 1><<<gemm_blocks(MROWS, 80), 128, 0, stream>>>(
            xih, DINNER, w_xp, DINNER, xdbch, 80,
            MROWS, 80, DINNER, nullptr, nullptr, 0);

        // dt_proj + bias + softplus: dtb[M,1536](f32) = sp(xdbch[:,:48] * w_dt[1536,48]^T + dpb)
        gemm_wmma_kernel<1, 0><<<gemm_blocks(MROWS, DINNER), 128, 0, stream>>>(
            xdbch, 80, w_dt, DTRANK, dtb, DINNER,
            MROWS, DINNER, DTRANK, dpb, nullptr, 0);

        // selective scan + skip + gate -> ybufh (f16)
        scan_kernel<<<dim3(DINNER / 256, BATCH), 256, 0, stream>>>(
            xdbch, dtb, xih, xz, al, dp, ybufh);

        // out_proj + residual: h = h + ybufh[M,1536] * w_op[768,1536]^T
        gemm_wmma_kernel<2, 0><<<gemm_blocks(MROWS, DMODEL), 128, 0, stream>>>(
            ybufh, DINNER, w_op, DINNER, h, DMODEL,
            MROWS, DMODEL, DINNER, nullptr, h, DMODEL);
    }

    // 3) Final norm, pooling, classifier
    rmsnorm_f32_kernel<<<MROWS, 256, 0, stream>>>(h, norm_f_w, hn);
    pool_kernel<<<(BATCH * DMODEL + 255) / 256, 256, 0, stream>>>(hn, mask, pooled);
    cls_kernel<<<1, 32, 0, stream>>>(pooled, cls_w, cls_b, logits);

    (void)in_sizes; (void)n_in; (void)out_size; (void)ws_size;
}
